// LocalSwarmAggregator_4758823764522
// MI455X (gfx1250) — matrix operand order repaired and checked by reading, not compiled, not run on hardware
//
#include <hip/hip_runtime.h>

typedef __bf16 bf16;
typedef __attribute__((ext_vector_type(16))) __bf16 v16bf;
typedef __attribute__((ext_vector_type(8)))  float  v8f;

#define D_MODEL 1024
#define NSEQ    2048
#define NBATCH  2
#define NHEAD   16
#define HD      64
#define MROWS   (NBATCH * NSEQ)   /* 4096 */
#define WINDOW  256
#define NEGBIG  (-1.0e9f)
#define LDP     40                /* padded LDS row stride in bf16 (80B, 16B-aligned, bank-spread) */
#define SCALE2  0.1803368801111204f   /* (1/sqrt(64)) * log2(e) : softmax done in exp2 domain */

// Load a 16-element bf16 fragment as two aligned 16-byte chunks.
static __device__ __forceinline__ v16bf ld2x16(const bf16* p0, const bf16* p1) {
    v16bf r;
    uint4* q = reinterpret_cast<uint4*>(&r);
    q[0] = *reinterpret_cast<const uint4*>(p0);
    q[1] = *reinterpret_cast<const uint4*>(p1);
    return r;
}

static __device__ __forceinline__ v8f wmma_bf16(v16bf a, v16bf b, v8f c) {
    return __builtin_amdgcn_wmma_f32_16x16x32_bf16(false, a, false, b, (short)0, c, false, false);
}

// ---- 16-lane-group XOR butterflies on the VALU (v_permlane16_b32), no LDS traffic ----
static __device__ __forceinline__ float permx(float v, int lo, int hi, int mask) {
#if __has_builtin(__builtin_amdgcn_permlane16)
    int i = __builtin_bit_cast(int, v);
    i = __builtin_amdgcn_permlane16(i, i, lo, hi, false, false);
    return __builtin_bit_cast(float, i);
#else
    return __shfl_xor(v, mask, 32);
#endif
}
static __device__ __forceinline__ float redmax16(float v) {
    v = fmaxf(v, permx(v, 0x67452301, (int)0xEFCDAB89, 1));
    v = fmaxf(v, permx(v, 0x54761032, (int)0xDCFE98BA, 2));
    v = fmaxf(v, permx(v, 0x32107654, (int)0xBA98FEDC, 4));
    v = fmaxf(v, permx(v, (int)0xFEDCBA98, 0x76543210, 8));
    return v;
}
static __device__ __forceinline__ float redsum16(float v) {
    v += permx(v, 0x67452301, (int)0xEFCDAB89, 1);
    v += permx(v, 0x54761032, (int)0xDCFE98BA, 2);
    v += permx(v, 0x32107654, (int)0xBA98FEDC, 4);
    v += permx(v, (int)0xFEDCBA98, 0x76543210, 8);
    return v;
}

// ---------------------------------------------------------------------------
// fp32 -> bf16 conversion (4 elements / thread, vectorized)
// ---------------------------------------------------------------------------
__global__ __launch_bounds__(256) void cvt_f32_bf16_kernel(const float* __restrict__ in,
                                                           bf16* __restrict__ out, int n4) {
    int i = blockIdx.x * 256 + threadIdx.x;
    if (i < n4) {
        float4 f = reinterpret_cast<const float4*>(in)[i];
        bf16 o[4] = {(bf16)f.x, (bf16)f.y, (bf16)f.z, (bf16)f.w};
        reinterpret_cast<uint2*>(out)[i] = *reinterpret_cast<uint2*>(o);
    }
}

// ---------------------------------------------------------------------------
// fp32 [K,N] -> bf16 transposed [N,K]  (32x32 LDS tile)
// ---------------------------------------------------------------------------
__global__ __launch_bounds__(256) void cvtT_f32_bf16_kernel(const float* __restrict__ in,
                                                            bf16* __restrict__ out) {
    __shared__ float tile[32][33];
    const int n0 = blockIdx.x * 32;
    const int k0 = blockIdx.y * 32;
    const int c  = threadIdx.x & 31;
    const int r0 = threadIdx.x >> 5;   // 0..7
#pragma unroll
    for (int rr = 0; rr < 4; ++rr) {
        int r = r0 + rr * 8;
        tile[r][c] = in[(size_t)(k0 + r) * D_MODEL + n0 + c];
    }
    __syncthreads();
#pragma unroll
    for (int rr = 0; rr < 4; ++rr) {
        int r = r0 + rr * 8;
        out[(size_t)(n0 + r) * D_MODEL + k0 + c] = (bf16)tile[c][r];
    }
}

// ---------------------------------------------------------------------------
// GEMM: C[4096,1024] = A[4096,1024](bf16) * Wt[1024,1024]^T(bf16, [N][K]) + bias
// LAYOUT 0: float out [M,N]; 1: bf16 [B,H,N,64]; 2: bf16 [B,H,64,N]
// 256 threads / 8 waves, 128x128 tile, K in 32-slabs, double-buffered LDS.
// ---------------------------------------------------------------------------
template <int LAYOUT, typename OT>
__global__ __launch_bounds__(256) void gemm_bf16_kernel(const bf16* __restrict__ A,
                                                        const bf16* __restrict__ Wt,
                                                        const float* __restrict__ bias,
                                                        OT* __restrict__ out) {
    __shared__ __align__(16) bf16 As[2][128 * LDP];
    __shared__ __align__(16) bf16 Bs[2][128 * LDP];   // transposed: [col][k]

    const int K = D_MODEL, NC = D_MODEL;
    const int n0 = blockIdx.x * 128;
    const int m0 = blockIdx.y * 128;
    const int t = threadIdx.x;
    const int wave = t >> 5, lane = t & 31;
    const int lh = lane & 15;
    const bool hi = lane >= 16;
    const int wm = wave >> 2, wn = wave & 3;
    const int klo  = hi ? 8  : 0;
    const int klo2 = hi ? 16 : 0;

    v8f acc[4][2];
#pragma unroll
    for (int i = 0; i < 4; ++i)
#pragma unroll
        for (int j = 0; j < 2; ++j)
#pragma unroll
            for (int r = 0; r < 8; ++r) acc[i][j][r] = 0.0f;

    const int srow = t >> 1, skk = (t & 1) * 16;
    const bf16* ag = A  + (size_t)(m0 + srow) * K + skk;
    const bf16* bg = Wt + (size_t)(n0 + srow) * K + skk;

    uint4 pa0, pa1, pb0, pb1;
    auto ldg = [&](int k0) {
        const uint4* pa = reinterpret_cast<const uint4*>(ag + k0);
        const uint4* pb = reinterpret_cast<const uint4*>(bg + k0);
        pa0 = pa[0]; pa1 = pa[1];
        pb0 = pb[0]; pb1 = pb[1];
    };
    auto sts = [&](int buf) {
        *reinterpret_cast<uint4*>(&As[buf][srow * LDP + skk])     = pa0;
        *reinterpret_cast<uint4*>(&As[buf][srow * LDP + skk + 8]) = pa1;
        *reinterpret_cast<uint4*>(&Bs[buf][srow * LDP + skk])     = pb0;
        *reinterpret_cast<uint4*>(&Bs[buf][srow * LDP + skk + 8]) = pb1;
    };
    auto compute = [&](int buf) {
        v16bf a[4], b[2];
#pragma unroll
        for (int i = 0; i < 4; ++i) {
            const bf16* p = &As[buf][(wm * 64 + i * 16 + lh) * LDP];
            a[i] = ld2x16(p + klo2, p + klo2 + 8);
        }
#pragma unroll
        for (int j = 0; j < 2; ++j) {
            const bf16* p = &Bs[buf][(wn * 32 + j * 16 + lh) * LDP + klo2];
            b[j] = ld2x16(p, p + 8);
        }
#pragma unroll
        for (int i = 0; i < 4; ++i)
#pragma unroll
            for (int j = 0; j < 2; ++j)
                acc[i][j] = wmma_bf16(a[i], b[j], acc[i][j]);
    };

    ldg(0);
    sts(0);
    __syncthreads();
    for (int step = 0; step < 31; ++step) {
        ldg((step + 1) * 32);     // prefetch next slab while computing
        compute(step & 1);
        sts((step + 1) & 1);
        __syncthreads();
    }
    compute(31 & 1);

    // Epilogue: C layout => VGPR r holds row r (lanes 0-15) / r+8 (lanes 16-31).
#pragma unroll
    for (int i = 0; i < 4; ++i) {
#pragma unroll
        for (int j = 0; j < 2; ++j) {
            int col = n0 + wn * 32 + j * 16 + lh;
            float bv = bias[col];
            int rbase = m0 + wm * 64 + i * 16 + (hi ? 8 : 0);
#pragma unroll
            for (int r = 0; r < 8; ++r) {
                float v = acc[i][j][r] + bv;
                int row = rbase + r;
                if (LAYOUT == 0) {
                    out[(size_t)row * NC + col] = (OT)v;
                } else {
                    int bb = row >> 11, n = row & (NSEQ - 1);
                    int h = col >> 6, d = col & 63;
                    size_t off;
                    if (LAYOUT == 1)
                        off = ((size_t)(bb * NHEAD + h) * NSEQ + n) * HD + d;
                    else
                        off = ((size_t)(bb * NHEAD + h) * HD + d) * NSEQ + n;
                    out[off] = (OT)v;
                }
            }
        }
    }
}

// ---------------------------------------------------------------------------
// Flash attention, causal + sliding-window(256).
// Q,K: [B*H, N, 64] bf16; V: [B*H, 64, N] bf16 (transposed); ctx: [B, N, D] bf16.
// 1 wave = 16 query rows; 4 waves / block; keys 32 at a time.
// ---------------------------------------------------------------------------
__global__ __launch_bounds__(128) void attn_kernel(const bf16* __restrict__ Q,
                                                   const bf16* __restrict__ Kh,
                                                   const bf16* __restrict__ Vt,
                                                   bf16* __restrict__ ctx) {
    __shared__ __align__(16) bf16 plds[4][16 * LDP];  // per-wave P tile (16 x 32)

    const int wave = threadIdx.x >> 5, lane = threadIdx.x & 31;
    const int lh = lane & 15;
    const bool hi = lane >= 16;
    const int gw = blockIdx.x * 4 + wave;
    const int qt = gw & 127;
    const int bh = gw >> 7;
    const int q0 = qt * 16;

    const bf16* q  = Q  + (size_t)bh * NSEQ * HD;
    const bf16* kp = Kh + (size_t)bh * NSEQ * HD;
    const bf16* vp = Vt + (size_t)bh * HD * NSEQ;

    const int klo = hi ? 8 : 0;
    const int fof = hi ? 16 : 0;
    const bf16* qrow = q + (size_t)(q0 + lh) * HD;
    v16bf aq0 = ld2x16(qrow + fof,      qrow + fof + 8);
    v16bf aq1 = ld2x16(qrow + 32 + fof, qrow + 32 + fof + 8);

    float m[8], l[8];
    v8f acc[4];
#pragma unroll
    for (int r = 0; r < 8; ++r) { m[r] = -3.0e38f; l[r] = 0.0f; }
#pragma unroll
    for (int c = 0; c < 4; ++c)
#pragma unroll
        for (int r = 0; r < 8; ++r) acc[c][r] = 0.0f;

    const int rowi = q0 + (hi ? 8 : 0);
    const int jstart = (q0 > WINDOW) ? ((q0 - WINDOW) & ~31) : 0;

    for (int kb = jstart; kb <= q0 + 15; kb += 32) {
        // ---- issue ALL global fragment loads for this key tile up front ----
        const bf16* kr  = kp + (size_t)(kb + lh) * HD + fof;
        const bf16* kr2 = kr + 16 * HD;
        v16bf b00 = ld2x16(kr,       kr + 8);        // keys kb..kb+15, feat 0..31
        v16bf b01 = ld2x16(kr + 32,  kr + 40);       // feat 32..63
        v16bf b10 = ld2x16(kr2,      kr2 + 8);       // keys kb+16..kb+31
        v16bf b11 = ld2x16(kr2 + 32, kr2 + 40);
        v16bf vb[4];
        const bf16* vbp = vp + kb + fof;
#pragma unroll
        for (int c = 0; c < 4; ++c) {
            const bf16* vcol = vbp + (size_t)(c * 16 + lh) * NSEQ;
            vb[c] = ld2x16(vcol, vcol + 8);
        }

        // ---- S = Q * K^T ----
        v8f s0 = {0.f, 0.f, 0.f, 0.f, 0.f, 0.f, 0.f, 0.f};
        v8f s1 = {0.f, 0.f, 0.f, 0.f, 0.f, 0.f, 0.f, 0.f};
        s0 = wmma_bf16(aq0, b00, s0);
        s0 = wmma_bf16(aq1, b01, s0);
        s1 = wmma_bf16(aq0, b10, s1);
        s1 = wmma_bf16(aq1, b11, s1);

        // ---- scale + mask + online softmax (exp2 domain, fp32) ----
        const int j0 = kb + lh, j1 = kb + 16 + lh;
        float alpha[8], p0v[8], p1v[8];
#pragma unroll
        for (int r = 0; r < 8; ++r) {
            int i = rowi + r;
            float x0 = s0[r] * SCALE2 + (((j0 > i) || (j0 < i - WINDOW)) ? NEGBIG : 0.0f);
            float x1 = s1[r] * SCALE2 + (((j1 > i) || (j1 < i - WINDOW)) ? NEGBIG : 0.0f);
            float mx = redmax16(fmaxf(x0, x1));
            float mn = fmaxf(m[r], mx);
            alpha[r] = __builtin_exp2f(m[r] - mn);
            float e0 = __builtin_exp2f(x0 - mn);
            float e1 = __builtin_exp2f(x1 - mn);
            l[r] = l[r] * alpha[r] + redsum16(e0 + e1);
            m[r] = mn;
            p0v[r] = e0; p1v[r] = e1;
        }

        // ---- re-layout P (C-layout) into an A-fragment via per-wave LDS ----
        bf16* pl = plds[wave];
#pragma unroll
        for (int r = 0; r < 8; ++r) {
            int prow = r + (hi ? 8 : 0);
            pl[prow * LDP + lh]      = (bf16)p0v[r];
            pl[prow * LDP + 16 + lh] = (bf16)p1v[r];
        }
        __builtin_amdgcn_wave_barrier();
        const bf16* pr = pl + lh * LDP + fof;
        v16bf pa = ld2x16(pr, pr + 8);

        // ---- rescale accumulators, then acc += P * V ----
#pragma unroll
        for (int c = 0; c < 4; ++c)
#pragma unroll
            for (int r = 0; r < 8; ++r) acc[c][r] *= alpha[r];
#pragma unroll
        for (int c = 0; c < 4; ++c)
            acc[c] = wmma_bf16(pa, vb[c], acc[c]);
    }

    // ---- normalize and store context [B, N, D] bf16 ----
    const int b = bh >> 4, h = bh & 15;
#pragma unroll
    for (int r = 0; r < 8; ++r) {
        float inv = __builtin_amdgcn_rcpf(l[r]);
        int tok = q0 + r + (hi ? 8 : 0);
#pragma unroll
        for (int c = 0; c < 4; ++c) {
            int col = h * HD + c * 16 + lh;
            ctx[(size_t)(b * NSEQ + tok) * D_MODEL + col] = (bf16)(acc[c][r] * inv);
        }
    }
}

// ---------------------------------------------------------------------------
// Host-side orchestration
// ---------------------------------------------------------------------------
extern "C" void kernel_launch(void* const* d_in, const int* in_sizes, int n_in,
                              void* d_out, int out_size, void* d_ws, size_t ws_size,
                              hipStream_t stream) {
    (void)in_sizes; (void)n_in; (void)out_size; (void)ws_size;

    const float* x  = (const float*)d_in[0];
    const float* Wq = (const float*)d_in[1];
    const float* bq = (const float*)d_in[2];
    const float* Wk = (const float*)d_in[3];
    const float* bk = (const float*)d_in[4];
    const float* Wv = (const float*)d_in[5];
    const float* bv = (const float*)d_in[6];
    const float* Wo = (const float*)d_in[7];
    const float* bo = (const float*)d_in[8];
    float* out = (float*)d_out;

    const size_t nX = (size_t)MROWS * D_MODEL;
    const size_t nW = (size_t)D_MODEL * D_MODEL;

    char* ws = (char*)d_ws;
    size_t o = 0;
    bf16* xb  = (bf16*)(ws + o); o += nX * sizeof(bf16);
    bf16* wqt = (bf16*)(ws + o); o += nW * sizeof(bf16);
    bf16* wkt = (bf16*)(ws + o); o += nW * sizeof(bf16);
    bf16* wvt = (bf16*)(ws + o); o += nW * sizeof(bf16);
    bf16* wot = (bf16*)(ws + o); o += nW * sizeof(bf16);
    bf16* Qb  = (bf16*)(ws + o); o += nX * sizeof(bf16);
    bf16* Kb  = (bf16*)(ws + o); o += nX * sizeof(bf16);
    bf16* Vb  = (bf16*)(ws + o); o += nX * sizeof(bf16);
    bf16* Cb  = (bf16*)(ws + o); o += nX * sizeof(bf16);

    // 1) conversions: x -> bf16; weights -> bf16 transposed [N][K]
    cvt_f32_bf16_kernel<<<(int)((nX / 4 + 255) / 256), 256, 0, stream>>>(x, xb, (int)(nX / 4));
    dim3 tgrid(D_MODEL / 32, D_MODEL / 32);
    cvtT_f32_bf16_kernel<<<tgrid, 256, 0, stream>>>(Wq, wqt);
    cvtT_f32_bf16_kernel<<<tgrid, 256, 0, stream>>>(Wk, wkt);
    cvtT_f32_bf16_kernel<<<tgrid, 256, 0, stream>>>(Wv, wvt);
    cvtT_f32_bf16_kernel<<<tgrid, 256, 0, stream>>>(Wo, wot);

    // 2) QKV projections (Q,K head layout; V head-transposed)
    dim3 ggrid(D_MODEL / 128, MROWS / 128);
    gemm_bf16_kernel<1, bf16><<<ggrid, 256, 0, stream>>>(xb, wqt, bq, Qb);
    gemm_bf16_kernel<1, bf16><<<ggrid, 256, 0, stream>>>(xb, wkt, bk, Kb);
    gemm_bf16_kernel<2, bf16><<<ggrid, 256, 0, stream>>>(xb, wvt, bv, Vb);

    // 3) sliding-window flash attention
    attn_kernel<<<(NBATCH * NHEAD * (NSEQ / 16)) / 4, 128, 0, stream>>>(Qb, Kb, Vb, Cb);

    // 4) output projection -> fp32
    gemm_bf16_kernel<0, float><<<ggrid, 256, 0, stream>>>(Cb, wot, bo, out);
}
